// Model_88330297409595
// MI455X (gfx1250) — matrix operand order repaired and checked by reading, not compiled, not run on hardware
//
#include <hip/hip_runtime.h>
#include <hip/hip_bf16.h>

// ---------------------------------------------------------------------------
// Model dims
// ---------------------------------------------------------------------------
#define BB     64
#define TT     384
#define PP     96
#define AA     3
#define MM     (BB*TT)        // 24576 rows
#define DIM    768
#define HEADS  128
#define HD     (DIM/HEADS)    // 6
#define WW     5
#define C0     (2*PP*AA)      // 576
#define D2     (2*DIM)        // 1536
#define NOUT   60
#define NOUTP  64             // padded to WMMA tile

typedef __attribute__((ext_vector_type(16))) _Float16 v16h;
typedef __attribute__((ext_vector_type(8)))  float    v8f;

// ---------------------------------------------------------------------------
// Weight convert + transpose: src fp32 [K,N] row-major -> dst f16 [Npad,K]
// ---------------------------------------------------------------------------
__global__ __launch_bounds__(256) void k_wt(const float* __restrict__ src,
                                            _Float16* __restrict__ dst,
                                            int K, int N, int Npad) {
  int idx = blockIdx.x * 256 + threadIdx.x;
  if (idx >= Npad * K) return;
  int n = idx / K, k = idx % K;
  dst[idx] = (n < N) ? (_Float16)src[(size_t)k * N + n] : (_Float16)0.f;
}

// ---------------------------------------------------------------------------
// mask: keep[m] = 1 if any of 288 entries nonzero else 0
// ---------------------------------------------------------------------------
__global__ __launch_bounds__(256) void k_mask(const float* __restrict__ x,
                                              float* __restrict__ keep) {
  __shared__ int flag;
  if (threadIdx.x == 0) flag = 0;
  __syncthreads();
  const float* row = x + (size_t)blockIdx.x * (PP * AA);
  bool any = false;
  for (int i = threadIdx.x; i < PP * AA; i += 256) any |= (row[i] != 0.f);
  if (any) flag = 1;
  __syncthreads();
  if (threadIdx.x == 0) keep[blockIdx.x] = flag ? 1.f : 0.f;
}

// ---------------------------------------------------------------------------
// Global masked stats over (T,P) per (b,a):  stats[(b*3+a)*2] = {mean, std}
// ---------------------------------------------------------------------------
__global__ __launch_bounds__(256) void k_stats(const float* __restrict__ x,
                                               float* __restrict__ stats) {
  int b = blockIdx.x / AA, a = blockIdx.x % AA;
  const float* xb = x + (size_t)b * TT * PP * AA + a;
  float sw = 0, sx = 0, sx2 = 0;
  for (int j = threadIdx.x; j < TT * PP; j += 256) {
    float v = xb[(size_t)j * AA];
    float w = (v != 0.f) ? 1.f : 0.f;
    sw += w; sx += v * w; sx2 += v * v * w;
  }
  __shared__ float sh[3][8];
  for (int off = 16; off; off >>= 1) {
    sw += __shfl_down(sw, off, 32);
    sx += __shfl_down(sx, off, 32);
    sx2 += __shfl_down(sx2, off, 32);
  }
  if ((threadIdx.x & 31) == 0) {
    int w5 = threadIdx.x >> 5;
    sh[0][w5] = sw; sh[1][w5] = sx; sh[2][w5] = sx2;
  }
  __syncthreads();
  if (threadIdx.x == 0) {
    float tw = 0, tx = 0, tx2 = 0;
    for (int i = 0; i < 8; ++i) { tw += sh[0][i]; tx += sh[1][i]; tx2 += sh[2][i]; }
    float denom = tw + 1e-5f;
    float mean  = tx / denom;
    float varn  = tx2 - 2.f * mean * tx + mean * mean * tw;
    stats[blockIdx.x * 2]     = mean;
    stats[blockIdx.x * 2 + 1] = sqrtf(varn / denom);
  }
}

// ---------------------------------------------------------------------------
// Front-end: build h0 f16 [MM, 576] = concat(global axis-LN, per-range LN)
// ---------------------------------------------------------------------------
__global__ __launch_bounds__(256) void k_h0(const float* __restrict__ x,
    const float* __restrict__ stats,
    const float* __restrict__ xg, const float* __restrict__ xb,
    const float* __restrict__ fg0, const float* __restrict__ fg1,
    const float* __restrict__ fg2, const float* __restrict__ fg3,
    const float* __restrict__ fb0, const float* __restrict__ fb1,
    const float* __restrict__ fb2, const float* __restrict__ fb3,
    _Float16* __restrict__ h0) {
  const int RS[5] = {0, 21, 42, 82, 96};
  int m = blockIdx.x, b = m / TT;
  __shared__ float xs[PP * AA];
  __shared__ float gm[12], gs[12];
  const float* row = x + (size_t)m * (PP * AA);
  for (int i = threadIdx.x; i < PP * AA; i += 256) xs[i] = row[i];
  __syncthreads();
  if (threadIdx.x < 12) {
    int ri = threadIdx.x / 3, a = threadIdx.x % 3;
    float sw = 0, sx = 0, sx2 = 0;
    for (int p = RS[ri]; p < RS[ri + 1]; ++p) {
      float v = xs[p * 3 + a];
      float w = (v != 0.f) ? 1.f : 0.f;
      sw += w; sx += v * w; sx2 += v * v * w;
    }
    float denom = sw + 1e-5f;
    float mean  = sx / denom;
    float varn  = sx2 - 2.f * mean * sx + mean * mean * sw;
    gm[threadIdx.x] = mean;
    gs[threadIdx.x] = sqrtf(varn / denom);
  }
  __syncthreads();
  _Float16* out = h0 + (size_t)m * C0;
  for (int j = threadIdx.x; j < PP * AA; j += 256) {
    int p = j / 3, a = j % 3;
    float v = xs[j];
    float w = (v != 0.f) ? 1.f : 0.f;
    float mean = stats[(b * 3 + a) * 2], stdv = stats[(b * 3 + a) * 2 + 1];
    float nx = (xg[j] * ((v - mean) / (stdv + 1e-5f)) + xb[j]) * w;
    int ri = p < 21 ? 0 : p < 42 ? 1 : p < 82 ? 2 : 3;
    int g = ri * 3 + a;
    const float* fg = ri == 0 ? fg0 : ri == 1 ? fg1 : ri == 2 ? fg2 : fg3;
    const float* fb = ri == 0 ? fb0 : ri == 1 ? fb1 : ri == 2 ? fb2 : fb3;
    int loc = (p - RS[ri]) * 3 + a;
    float ft = (fg[loc] * ((v - gm[g]) / (gs[g] + 1e-5f)) + fb[loc]) * w;
    out[j]            = (_Float16)nx;   // nx block: cols 0..287
    out[PP * AA + j]  = (_Float16)ft;   // feats block: cols 288..575
  }
}

// ---------------------------------------------------------------------------
// Positional encoding [TT, DIM]
// ---------------------------------------------------------------------------
__global__ __launch_bounds__(256) void k_posenc(float* __restrict__ pe) {
  int idx = blockIdx.x * 256 + threadIdx.x;
  if (idx >= TT * DIM) return;
  int t = idx / DIM, i = idx % DIM;
  int k = i >> 1;
  float div = __expf((float)(2 * k) * (-9.210340371976184f / (float)DIM));
  float ang = (float)t * div;
  pe[idx] = (i & 1) ? __cosf(ang) : __sinf(ang);
}

// ---------------------------------------------------------------------------
// WMMA GEMM: C[M,N] (f32) = A[M,K] (f16, row-major) @ Wt[N,K]^T (f16)
// Register-blocked: one 64x64 tile per wave = 4x4 accumulators of 16x16.
// Per K-step of 32: 4 A-fragments + 4 B-fragments feed 16 WMMAs
// (load:wmma ratio 1:1, 4x less L2 traffic than 1 tile/wave).
// Fragment lane layouts per CDNA5 ISA 7.12.2.
// ---------------------------------------------------------------------------
__global__ __launch_bounds__(256) void k_gemm(const _Float16* __restrict__ A,
                                              const _Float16* __restrict__ Wt,
                                              float* __restrict__ C,
                                              int M, int K, int N, int tiles) {
  int wave = threadIdx.x >> 5;
  int lane = threadIdx.x & 31;
  int tile = blockIdx.x * 8 + wave;
  if (tile >= tiles) return;
  int nt64 = N >> 6;
  int mt = tile / nt64;          // 64-row block
  int nt = tile % nt64;          // 64-col block
  int hi  = (lane & 16) ? 1 : 0;
  int l15 = lane & 15;
  // A lane row m = base + (lane&15); lo-half lanes hold K {0..7,16..23}+kk,
  // hi-half lanes hold K {8..15,24..31}+kk.
  const _Float16* arow[4];
  const _Float16* brow[4];
#pragma unroll
  for (int i = 0; i < 4; ++i)
    arow[i] = A + (size_t)(mt * 64 + i * 16 + l15) * K + (hi ? 8 : 0);
  // B lane col n = base + (lane&15); lo-half lanes K {0..7,16..23}+kk, hi K {8..15,24..31}+kk
#pragma unroll
  for (int j = 0; j < 4; ++j)
    brow[j] = Wt + (size_t)(nt * 64 + j * 16 + l15) * K + (hi ? 8 : 0);

  v8f acc[4][4];
#pragma unroll
  for (int i = 0; i < 4; ++i)
#pragma unroll
    for (int j = 0; j < 4; ++j) acc[i][j] = (v8f){};

  for (int kk = 0; kk < K; kk += 32) {
    v16h af[4], bf[4];
#pragma unroll
    for (int i = 0; i < 4; ++i) {
      __builtin_prefetch(arow[i] + kk + 128, 0, 1);
      ((uint4*)&af[i])[0] = *(const uint4*)(arow[i] + kk);
      ((uint4*)&af[i])[1] = *(const uint4*)(arow[i] + kk + 16);
    }
#pragma unroll
    for (int j = 0; j < 4; ++j) {
      __builtin_prefetch(brow[j] + kk + 128, 0, 1);
      ((uint4*)&bf[j])[0] = *(const uint4*)(brow[j] + kk);
      ((uint4*)&bf[j])[1] = *(const uint4*)(brow[j] + kk + 16);
    }
#pragma unroll
    for (int i = 0; i < 4; ++i)
#pragma unroll
      for (int j = 0; j < 4; ++j)
        acc[i][j] = __builtin_amdgcn_wmma_f32_16x16x32_f16(
            false, af[i], false, bf[j], (short)0, acc[i][j], false, false);
  }
  // C/D layout: VGPR r -> row r (lanes 0-15) / row 8+r (lanes 16-31), col lane&15
#pragma unroll
  for (int i = 0; i < 4; ++i) {
#pragma unroll
    for (int j = 0; j < 4; ++j) {
      float* crow = C + (size_t)(mt * 64 + i * 16 + (hi ? 8 : 0)) * N
                      + nt * 64 + j * 16 + l15;
#pragma unroll
      for (int r = 0; r < 8; ++r) crow[(size_t)r * N] = acc[i][j][r];
    }
  }
}

// ---------------------------------------------------------------------------
// bias + LayerNorm + ELU (+ optional residual/posenc), dual f32/f16 output.
// addmode: 0 none, 1 full residual [M,N], 2 posenc [TT,N] (row m%TT)
// ---------------------------------------------------------------------------
__global__ __launch_bounds__(256) void k_ln_elu(const float* __restrict__ Z,
    const float* __restrict__ bias, const float* __restrict__ g,
    const float* __restrict__ b2, const float* __restrict__ addp, int addmode,
    float* __restrict__ outF, _Float16* __restrict__ outH, int N) {
  int m = blockIdx.x;
  const float* z = Z + (size_t)m * N;
  float s = 0, s2 = 0;
  for (int n = threadIdx.x; n < N; n += 256) {
    float v = z[n] + bias[n];
    s += v; s2 += v * v;
  }
  __shared__ float red[16];
  __shared__ float mv[2];
  for (int off = 16; off; off >>= 1) {
    s += __shfl_down(s, off, 32);
    s2 += __shfl_down(s2, off, 32);
  }
  if ((threadIdx.x & 31) == 0) {
    red[threadIdx.x >> 5] = s;
    red[8 + (threadIdx.x >> 5)] = s2;
  }
  __syncthreads();
  if (threadIdx.x == 0) {
    float ts = 0, ts2 = 0;
    for (int i = 0; i < 8; ++i) { ts += red[i]; ts2 += red[8 + i]; }
    float mean = ts / (float)N;
    float var  = ts2 / (float)N - mean * mean;
    mv[0] = mean;
    mv[1] = rsqrtf(var + 1e-5f);
  }
  __syncthreads();
  float mean = mv[0], rstd = mv[1];
  const float* ap = addp + (addmode == 1 ? (size_t)m * N
                          : addmode == 2 ? (size_t)(m % TT) * N : 0);
  for (int n = threadIdx.x; n < N; n += 256) {
    float v = z[n] + bias[n];
    float o = g[n] * (v - mean) * rstd + b2[n];
    o = (o > 0.f) ? o : (__expf(o) - 1.f);
    if (addmode) o += ap[n];
    outF[(size_t)m * N + n] = o;
    outH[(size_t)m * N + n] = (_Float16)o;
  }
}

// ---------------------------------------------------------------------------
// Attention score epilogue: S = exp(S + b) * keep[m]   (in place, [MM,HEADS])
// ---------------------------------------------------------------------------
__global__ __launch_bounds__(256) void k_score(float* __restrict__ S,
                                               const float* __restrict__ ab,
                                               const float* __restrict__ keep) {
  int idx = blockIdx.x * 256 + threadIdx.x;
  int m = idx >> 7, h = idx & 127;
  S[idx] = __expf(S[idx] + ab[h]) * keep[m];
}

// ---------------------------------------------------------------------------
// Sliding-window attention combine: one thread per (b,t,head), 6 dims each.
// out (f16 [MM,DIM]) feeds o_w GEMM directly.
// ---------------------------------------------------------------------------
__global__ __launch_bounds__(256) void k_slide(const float* __restrict__ S,
    const float* __restrict__ V, const float* __restrict__ pos,
    _Float16* __restrict__ out, int dilation) {
  int idx = blockIdx.x * 256 + threadIdx.x;
  int h = idx & 127;
  int bt = idx >> 7;
  int t = bt % TT;
  int b = bt / TT;
  int base = (dilation * WW) / 2;
  float w[WW]; int tw[WW];
  float denom = 1e-5f;
#pragma unroll
  for (int i = 0; i < WW; ++i) {
    int p = t + dilation * i - base;
    tw[i] = p;
    float a = 0.f;
    if (p >= 0 && p < TT)
      a = S[((size_t)(b * TT + p)) * HEADS + h] * __expf(pos[i * HEADS + h]);
    w[i] = a; denom += a;
  }
  float inv = 1.f / denom;
  size_t obase = (size_t)bt * DIM + h * HD;
#pragma unroll
  for (int d = 0; d < HD; ++d) {
    float acc = 0.f;
#pragma unroll
    for (int i = 0; i < WW; ++i) {
      int p = tw[i];
      if (p >= 0 && p < TT && w[i] != 0.f)
        acc += w[i] * V[((size_t)(b * TT + p)) * DIM + h * HD + d];
    }
    out[obase + d] = (_Float16)(acc * inv);
  }
}

// ---------------------------------------------------------------------------
// Final: strip N-padding, add bias
// ---------------------------------------------------------------------------
__global__ __launch_bounds__(256) void k_final(const float* __restrict__ y64,
                                               const float* __restrict__ ob,
                                               float* __restrict__ out) {
  int idx = blockIdx.x * 256 + threadIdx.x;
  if (idx >= MM * NOUT) return;
  int m = idx / NOUT, n = idx % NOUT;
  out[idx] = y64[(size_t)m * NOUTP + n] + ob[n];
}

// ---------------------------------------------------------------------------
// Host side
// ---------------------------------------------------------------------------
static void gemm(const _Float16* A, const _Float16* Wt, float* C,
                 int M, int K, int N, hipStream_t s) {
  int tiles = (M / 64) * (N / 64);
  int blocks = (tiles + 7) / 8;
  k_gemm<<<blocks, 256, 0, s>>>(A, Wt, C, M, K, N, tiles);
}

static void conv_wt(const float* src, _Float16* dst, int K, int N, int Npad,
                    hipStream_t s) {
  int n = Npad * K;
  k_wt<<<(n + 255) / 256, 256, 0, s>>>(src, dst, K, N, Npad);
}

extern "C" void kernel_launch(void* const* d_in, const int* in_sizes, int n_in,
                              void* d_out, int out_size, void* d_ws, size_t ws_size,
                              hipStream_t stream) {
  (void)in_sizes; (void)n_in; (void)out_size; (void)ws_size;
  // Input order (leaf flatten of setup_inputs dict):
  // 0:x 1:xn_g 2:xn_b 3-6:fn_g[0..3] 7-10:fn_b[0..3]
  // 11:in1_w 12:in1_b 13:in1_g 14:in1_b2 15:in2_w 16:in2_b 17:in2_g 18:in2_b2
  // 19+11*l: attn_w, attn_b, pos, v_w, v_b, v_ln_g, v_ln_b, o_w, o_b, o_ln_g, o_ln_b
  // 52:out_w 53:out_b
  const float* x     = (const float*)d_in[0];
  const float* xn_g  = (const float*)d_in[1];
  const float* xn_b  = (const float*)d_in[2];
  const float* fg[4] = {(const float*)d_in[3], (const float*)d_in[4],
                        (const float*)d_in[5], (const float*)d_in[6]};
  const float* fb[4] = {(const float*)d_in[7], (const float*)d_in[8],
                        (const float*)d_in[9], (const float*)d_in[10]};
  const float* in1_w = (const float*)d_in[11];
  const float* in1_b = (const float*)d_in[12];
  const float* in1_g = (const float*)d_in[13];
  const float* in1_b2= (const float*)d_in[14];
  const float* in2_w = (const float*)d_in[15];
  const float* in2_b = (const float*)d_in[16];
  const float* in2_g = (const float*)d_in[17];
  const float* in2_b2= (const float*)d_in[18];
  const float* out_w = (const float*)d_in[52];
  const float* out_b = (const float*)d_in[53];

  char* ws = (char*)d_ws;
  size_t off = 0;
  auto alloc = [&](size_t bytes) -> char* {
    char* p = ws + off;
    off += (bytes + 255) & ~(size_t)255;
    return p;
  };
  _Float16* wt_in1 = (_Float16*)alloc((size_t)D2 * C0 * 2);
  _Float16* wt_in2 = (_Float16*)alloc((size_t)DIM * D2 * 2);
  _Float16* wt_attn[3], * wt_v[3], * wt_o[3];
  for (int l = 0; l < 3; ++l) {
    wt_attn[l] = (_Float16*)alloc((size_t)HEADS * DIM * 2);
    wt_v[l]    = (_Float16*)alloc((size_t)DIM * DIM * 2);
    wt_o[l]    = (_Float16*)alloc((size_t)DIM * DIM * 2);
  }
  _Float16* wt_out = (_Float16*)alloc((size_t)NOUTP * DIM * 2);
  float*    keep   = (float*)alloc((size_t)MM * 4);
  float*    stats  = (float*)alloc((size_t)BB * AA * 2 * 4);
  float*    pe     = (float*)alloc((size_t)TT * DIM * 4);
  _Float16* h0     = (_Float16*)alloc((size_t)MM * C0 * 2);
  float*    Z      = (float*)alloc((size_t)MM * D2 * 4);
  float*    H      = (float*)alloc((size_t)MM * DIM * 4);
  _Float16* Hb     = (_Float16*)alloc((size_t)MM * DIM * 2);
  char*     bufA   = alloc((size_t)MM * D2 * 2);  // H1b f16 [MM,D2] then V f32 [MM,DIM]
  _Float16* AOb    = (_Float16*)alloc((size_t)MM * DIM * 2);
  float*    S      = (float*)alloc((size_t)MM * HEADS * 4);
  float*    Y64    = (float*)alloc((size_t)MM * NOUTP * 4);

  // ---- one-time prep ----
  conv_wt(in1_w, wt_in1, C0, D2, D2, stream);
  conv_wt(in2_w, wt_in2, D2, DIM, DIM, stream);
  for (int l = 0; l < 3; ++l) {
    const int base = 19 + 11 * l;
    conv_wt((const float*)d_in[base + 0], wt_attn[l], DIM, HEADS, HEADS, stream);
    conv_wt((const float*)d_in[base + 3], wt_v[l],    DIM, DIM, DIM, stream);
    conv_wt((const float*)d_in[base + 7], wt_o[l],    DIM, DIM, DIM, stream);
  }
  conv_wt(out_w, wt_out, DIM, NOUT, NOUTP, stream);

  k_mask<<<MM, 256, 0, stream>>>(x, keep);
  k_stats<<<BB * AA, 256, 0, stream>>>(x, stats);
  k_h0<<<MM, 256, 0, stream>>>(x, stats, xn_g, xn_b,
                               fg[0], fg[1], fg[2], fg[3],
                               fb[0], fb[1], fb[2], fb[3], h0);
  k_posenc<<<(TT * DIM + 255) / 256, 256, 0, stream>>>(pe);

  // ---- input projections ----
  _Float16* H1b = (_Float16*)bufA;
  gemm(h0, wt_in1, Z, MM, C0, D2, stream);
  k_ln_elu<<<MM, 256, 0, stream>>>(Z, in1_b, in1_g, in1_b2, Z, 0, Z, H1b, D2);
  gemm(H1b, wt_in2, Z, MM, D2, DIM, stream);
  k_ln_elu<<<MM, 256, 0, stream>>>(Z, in2_b, in2_g, in2_b2, pe, 2, H, Hb, DIM);

  // ---- attention layers ----
  const int dil[3] = {1, 3, 3};
  for (int l = 0; l < 3; ++l) {
    const int base = 19 + 11 * l;
    const float* attn_b = (const float*)d_in[base + 1];
    const float* pos    = (const float*)d_in[base + 2];
    const float* v_b    = (const float*)d_in[base + 4];
    const float* v_g    = (const float*)d_in[base + 5];
    const float* v_bb   = (const float*)d_in[base + 6];
    const float* o_b    = (const float*)d_in[base + 8];
    const float* o_g    = (const float*)d_in[base + 9];
    const float* o_bb   = (const float*)d_in[base + 10];
    float* V = (float*)bufA;

    gemm(Hb, wt_attn[l], S, MM, DIM, HEADS, stream);
    k_score<<<MM * HEADS / 256, 256, 0, stream>>>(S, attn_b, keep);

    gemm(Hb, wt_v[l], Z, MM, DIM, DIM, stream);
    k_ln_elu<<<MM, 256, 0, stream>>>(Z, v_b, v_g, v_bb, Z, 0, V, Hb, DIM);

    k_slide<<<MM * HEADS / 256, 256, 0, stream>>>(S, V, pos, AOb, dil[l]);

    gemm(AOb, wt_o[l], Z, MM, DIM, DIM, stream);
    k_ln_elu<<<MM, 256, 0, stream>>>(Z, o_b, o_g, o_bb, H, 1, H, Hb, DIM);
  }

  // ---- output projection ----
  gemm(Hb, wt_out, Y64, MM, DIM, NOUTP, stream);
  k_final<<<(MM * NOUT + 255) / 256, 256, 0, stream>>>(Y64, out_b, (float*)d_out);
}
